// PaddedSoftmaxSHCSA_13443247636576
// MI455X (gfx1250) — matrix operand order repaired and checked by reading, not compiled, not run on hardware
//
#include <hip/hip_runtime.h>
#include <math.h>

// ---------------- problem constants ----------------
#define TT      4096          // tokens
#define CC      2048          // input channels
#define NF      2048          // per-head feature width (q/k/v)
#define NQKV    6144          // 3*NF
#define NEGBIG  (-1e30f)

// padded LDS row strides (in ushorts) to break bank conflicts
#define AROW    40            // 32 + 8   (kernel-1 A/B tiles)
#define QROW    2056          // 2048 + 8 (kernel-2 staged Q rows)
#define SROW    4104          // 4096 + 8 (kernel-2 score rows)

// ---------------- WMMA fragment types ----------------
typedef __attribute__((ext_vector_type(16))) __bf16 v16bf;
typedef __attribute__((ext_vector_type(8)))  float  v8f;

union FragBF {
    v16bf v;
    uint4 q[2];
};

static __device__ inline v8f zero8() {
    v8f z = {0.f,0.f,0.f,0.f,0.f,0.f,0.f,0.f};
    return z;
}

static __device__ inline unsigned short f32_to_bf16(float f) {
    unsigned int u = __builtin_bit_cast(unsigned int, f);
    unsigned int r = u + 0x7FFFu + ((u >> 16) & 1u);   // round-to-nearest-even
    return (unsigned short)(r >> 16);
}
static __device__ inline float bf16_to_f32(unsigned short h) {
    unsigned int u = ((unsigned int)h) << 16;
    return __builtin_bit_cast(float, u);
}

static __device__ inline v8f wmma_bf16(const FragBF& a, const FragBF& b, v8f c) {
    // D = A(16x32 bf16) * B(32x16 bf16) + C(16x16 f32)
    return __builtin_amdgcn_wmma_f32_16x16x32_bf16(
        /*neg_a=*/false, a.v, /*neg_b=*/false, b.v,
        /*c_mod=*/(short)0, c, /*reuse_a=*/false, /*reuse_b=*/false);
}

// =====================================================================
// Kernel 1: qkv = x @ W + b   (f32 in, bf16 out)
//   Q  : [TT][NF]  row-major bf16
//   Kb : [TT][NF]  row-major bf16
//   Vt : [NF][TT]  (transposed V) bf16  -> contiguous along token dim
// Block tile 128(M) x 128(N), 8 waves in 2x4, wave tile 64x32, K-step 32.
// =====================================================================
__global__ __launch_bounds__(256)
void qkv_gemm_kernel(const float* __restrict__ x,
                     const float* __restrict__ W,
                     const float* __restrict__ bias,
                     unsigned short* __restrict__ Q,
                     unsigned short* __restrict__ Kb,
                     unsigned short* __restrict__ Vt)
{
    __shared__ unsigned short As[128 * AROW];   // [row][k], padded
    __shared__ unsigned short Bs[128 * AROW];   // [col][k], padded (transposed)

    const int tid  = threadIdx.x;
    const int m0   = blockIdx.x * 128;
    const int n0   = blockIdx.y * 128;
    const int w    = tid >> 5;
    const int lane = tid & 31;
    const int lo   = lane & 15;
    const int hi   = lane >> 4;
    const int wm   = w & 1;        // 0..1 -> 64-row slab
    const int wn   = w >> 1;       // 0..3 -> 32-col slab

    v8f acc[4][2];
#pragma unroll
    for (int mi = 0; mi < 4; ++mi)
#pragma unroll
        for (int ni = 0; ni < 2; ++ni) acc[mi][ni] = zero8();

    for (int kk = 0; kk < CC; kk += 32) {
        // ---- cooperative global loads (f32) ----
        float4 av[4], bv[4];
#pragma unroll
        for (int j = 0; j < 4; ++j) {
            int f   = tid + 256 * j;           // 1024 float4 of A tile
            int row = f >> 3;                  // 8 float4 per 32-wide row
            int c4  = f & 7;
            av[j] = *(const float4*)(x + (m0 + row) * CC + kk + c4 * 4);
        }
#pragma unroll
        for (int j = 0; j < 4; ++j) {
            int f  = tid + 256 * j;            // 1024 float4 of B tile
            int kr = f >> 5;                   // 32 float4 per 128-wide row
            int c4 = f & 31;
            bv[j] = *(const float4*)(W + (kk + kr) * NQKV + n0 + c4 * 4);
        }

        __syncthreads();                       // prior compute done reading LDS
#pragma unroll
        for (int j = 0; j < 4; ++j) {
            int f   = tid + 256 * j;
            int row = f >> 3;
            int c4  = f & 7;
            unsigned short* p = &As[row * AROW + c4 * 4];
            p[0] = f32_to_bf16(av[j].x);
            p[1] = f32_to_bf16(av[j].y);
            p[2] = f32_to_bf16(av[j].z);
            p[3] = f32_to_bf16(av[j].w);
        }
#pragma unroll
        for (int j = 0; j < 4; ++j) {
            int f  = tid + 256 * j;
            int kr = f >> 5;
            int c4 = f & 31;
            Bs[(c4 * 4 + 0) * AROW + kr] = f32_to_bf16(bv[j].x);
            Bs[(c4 * 4 + 1) * AROW + kr] = f32_to_bf16(bv[j].y);
            Bs[(c4 * 4 + 2) * AROW + kr] = f32_to_bf16(bv[j].z);
            Bs[(c4 * 4 + 3) * AROW + kr] = f32_to_bf16(bv[j].w);
        }
        __syncthreads();

        // ---- fragments + WMMA ----
        FragBF a[4];
#pragma unroll
        for (int mi = 0; mi < 4; ++mi) {
            int arow = wm * 64 + mi * 16 + lo;
            const unsigned short* p = &As[arow * AROW + 16 * hi];
            a[mi].q[0] = *(const uint4*)(p);
            a[mi].q[1] = *(const uint4*)(p + 8);
        }
        FragBF bfr[2];
#pragma unroll
        for (int ni = 0; ni < 2; ++ni) {
            int bcol = wn * 32 + ni * 16 + lo;
            const unsigned short* p = &Bs[bcol * AROW + 16 * hi];
            bfr[ni].q[0] = *(const uint4*)(p);
            bfr[ni].q[1] = *(const uint4*)(p + 8);
        }
#pragma unroll
        for (int mi = 0; mi < 4; ++mi)
#pragma unroll
            for (int ni = 0; ni < 2; ++ni)
                acc[mi][ni] = wmma_bf16(a[mi], bfr[ni], acc[mi][ni]);
    }

    // ---- bias + store; region branch is BLOCK-UNIFORM (128 | 2048) ----
#define QKV_EPILOGUE(STORE_STMT)                                          \
    _Pragma("unroll")                                                     \
    for (int ni = 0; ni < 2; ++ni) {                                      \
        int gcol = n0 + wn * 32 + ni * 16 + lo;                           \
        float bv = bias[gcol];                                            \
        _Pragma("unroll")                                                 \
        for (int mi = 0; mi < 4; ++mi) {                                  \
            _Pragma("unroll")                                             \
            for (int i = 0; i < 8; ++i) {                                 \
                int grow = m0 + wm * 64 + mi * 16 + i + 8 * hi;           \
                unsigned short h = f32_to_bf16(acc[mi][ni][i] + bv);      \
                STORE_STMT;                                               \
            }                                                             \
        }                                                                 \
    }

    if (n0 < NF) {
        QKV_EPILOGUE(Q[grow * NF + gcol] = h)
    } else if (n0 < 2 * NF) {
        QKV_EPILOGUE(Kb[grow * NF + (gcol - NF)] = h)
    } else {
        QKV_EPILOGUE(Vt[(gcol - 2 * NF) * TT + grow] = h)
    }
#undef QKV_EPILOGUE
}

// =====================================================================
// Kernel 2: fused causal+padded attention for a 16-row block.
//   Stage:   Qs[16][2048] bf16 in LDS (64 KB, loaded once)
//   Phase 1: S[16][4096] = scale * q@k^T (masked), bf16 in LDS (128 KB);
//            wave w owns 128-col j-blocks w, w+8, ... (8 accums, A amortized)
//   Phase 2: row softmax (wave shfl reductions), padding rows -> 0
//   Phase 3: y = P @ Vt  (per-wave 16x256 f32 accumulators)
// =====================================================================
__global__ __launch_bounds__(256)
void attn_kernel(const unsigned short* __restrict__ Q,
                 const unsigned short* __restrict__ Kb,
                 const unsigned short* __restrict__ Vt,
                 const int* __restrict__ npad_p,
                 float* __restrict__ y)
{
    __shared__ unsigned short S [16 * SROW];   // ~128 KB score/prob buffer
    __shared__ unsigned short Qs[16 * QROW];   // ~64 KB staged q rows

    const int r0   = blockIdx.x * 16;
    const int tid  = threadIdx.x;
    const int w    = tid >> 5;
    const int lane = tid & 31;
    const int lo   = lane & 15;
    const int hi   = lane >> 4;
    const int npad = *npad_p;
    const float scale = 1.0f / sqrtf((float)NF);

    const int nj   = (r0 + 16 + 127) >> 7;    // causal: only j-blocks <= row block
    const int kend = nj << 7;                 // multiple of 128 (and of 32)

    // -------- Stage q row-block into LDS (16 x 2048 bf16, b128 loads) --------
#pragma unroll
    for (int j = 0; j < 16; ++j) {            // 4096 uint4 total, 16 per thread
        int f   = tid + 256 * j;
        int row = f >> 8;                     // 256 uint4 per row
        int c8  = f & 255;                    // 8 ushorts per uint4
        *(uint4*)&Qs[row * QROW + c8 * 8] =
            *(const uint4*)(Q + (r0 + row) * CC + c8 * 8);
    }
    __syncthreads();

    // -------- Phase 1: scores; wave w owns j-blocks w, w+8, w+16, ... --------
    for (int jb = w; jb < nj; jb += 8) {
        const int c0 = jb * 128;
        v8f acc[8];
#pragma unroll
        for (int ct = 0; ct < 8; ++ct) acc[ct] = zero8();

        for (int kk = 0; kk < CC; kk += 32) {
            FragBF a;                          // shared across 8 column tiles
            {
                const unsigned short* p = &Qs[lo * QROW + kk + 16 * hi];
                a.q[0] = *(const uint4*)(p);
                a.q[1] = *(const uint4*)(p + 8);
            }
#pragma unroll
            for (int ct = 0; ct < 8; ++ct) {
                const unsigned short* p = Kb + (c0 + ct * 16 + lo) * CC + kk + 16 * hi;
                FragBF b;
                b.q[0] = *(const uint4*)(p);
                b.q[1] = *(const uint4*)(p + 8);
                acc[ct] = wmma_bf16(a, b, acc[ct]);
            }
        }
#pragma unroll
        for (int ct = 0; ct < 8; ++ct) {
#pragma unroll
            for (int i = 0; i < 8; ++i) {
                int rl   = i + 8 * hi;
                int grow = r0 + rl;
                int gcol = c0 + ct * 16 + lo;
                bool ok  = (gcol <= grow) && (gcol >= npad) && (grow >= npad);
                float v  = ok ? acc[ct][i] * scale : NEGBIG;
                S[rl * SROW + gcol] = f32_to_bf16(v);
            }
        }
    }
    __syncthreads();

    // -------- Phase 2: row softmax (wave w handles rows 2w, 2w+1) --------
#pragma unroll
    for (int rr = 0; rr < 2; ++rr) {
        const int r    = w * 2 + rr;
        const int grow = r0 + r;
        float m = -3.0e38f;
        for (int c = lane; c < kend; c += 32)
            m = fmaxf(m, bf16_to_f32(S[r * SROW + c]));
#pragma unroll
        for (int off = 16; off > 0; off >>= 1)
            m = fmaxf(m, __shfl_xor(m, off, 32));
        float s = 0.f;
        for (int c = lane; c < kend; c += 32)
            s += __expf(bf16_to_f32(S[r * SROW + c]) - m);
#pragma unroll
        for (int off = 16; off > 0; off >>= 1)
            s += __shfl_xor(s, off, 32);
        float inv = (grow >= npad) ? (1.0f / s) : 0.0f;   // padding rows -> p = 0
        for (int c = lane; c < kend; c += 32) {
            float p = __expf(bf16_to_f32(S[r * SROW + c]) - m) * inv;
            S[r * SROW + c] = f32_to_bf16(p);
        }
    }
    __syncthreads();

    // -------- Phase 3: y = P @ V (via Vt), wave owns 16 x 256 of output --------
    const int ncol0 = w * 256;
    v8f acc2[16];
#pragma unroll
    for (int nt = 0; nt < 16; ++nt) acc2[nt] = zero8();

    for (int kk = 0; kk < kend; kk += 32) {
        FragBF a;
        {
            const unsigned short* p = &S[lo * SROW + kk + 16 * hi];
            a.q[0] = *(const uint4*)(p);
            a.q[1] = *(const uint4*)(p + 8);
        }
        if (kk + 32 < kend)
            __builtin_prefetch(Vt + (ncol0 + lo) * TT + kk + 32, 0, 1);
#pragma unroll
        for (int nt = 0; nt < 16; ++nt) {
            const unsigned short* p = Vt + (ncol0 + nt * 16 + lo) * TT + kk + 16 * hi;
            FragBF b;
            b.q[0] = *(const uint4*)(p);
            b.q[1] = *(const uint4*)(p + 8);
            acc2[nt] = wmma_bf16(a, b, acc2[nt]);
        }
    }

#pragma unroll
    for (int nt = 0; nt < 16; ++nt) {
        int gcol = ncol0 + nt * 16 + lo;
#pragma unroll
        for (int i = 0; i < 8; ++i) {
            int grow = r0 + i + 8 * hi;
            y[grow * NF + gcol] = acc2[nt][i];
        }
    }
}

// =====================================================================
extern "C" void kernel_launch(void* const* d_in, const int* in_sizes, int n_in,
                              void* d_out, int out_size, void* d_ws, size_t ws_size,
                              hipStream_t stream) {
    const float* x    = (const float*)d_in[0];
    const float* W    = (const float*)d_in[1];
    const float* bias = (const float*)d_in[2];
    const int*   npad = (const int*)d_in[3];

    unsigned short* Q  = (unsigned short*)d_ws;               // [TT][NF] bf16
    unsigned short* Kb = Q  + (size_t)TT * NF;                // [TT][NF] bf16
    unsigned short* Vt = Kb + (size_t)TT * NF;                // [NF][TT] bf16
    float* y = (float*)d_out;

    dim3 g1(TT / 128, NQKV / 128);
    qkv_gemm_kernel<<<g1, 256, 0, stream>>>(x, W, bias, Q, Kb, Vt);

    attn_kernel<<<TT / 16, 256, 0, stream>>>(Q, Kb, Vt, npad, y);
}
